// MoELayer_47193100648722
// MI455X (gfx1250) — matrix operand order repaired and checked by reading, not compiled, not run on hardware
//
#include <hip/hip_runtime.h>
#include <stdint.h>

#define N_TOK 8192
#define DDIM  2048
#define HDIM  2048
#define NEXP  8
#define BK    32    // bf16 WMMA K
#define BN    128   // fallback block N

typedef __attribute__((ext_vector_type(16))) __bf16 v16bf;
typedef __attribute__((ext_vector_type(8)))  float  v8f;

__device__ __forceinline__ uint16_t f2bf_bits(float f) {
  union { float f; uint32_t u; } c; c.f = f;
  uint32_t u = c.u + 0x7FFFu + ((c.u >> 16) & 1u);   // round-to-nearest-even
  return (uint16_t)(u >> 16);
}
__device__ __forceinline__ __bf16 f2bf(float f) {
  union { uint16_t s; __bf16 b; } r; r.s = f2bf_bits(f);
  return r.b;
}
__device__ __forceinline__ uint32_t pack2bf(float lo, float hi) {
  return (uint32_t)f2bf_bits(lo) | ((uint32_t)f2bf_bits(hi) << 16);
}

// A frag (ISA 7.12.2): VGPR0-3 = K [hi*8..hi*8+7], VGPR4-7 = +16
__device__ __forceinline__ v16bf load_frag_a(const __bf16* row, int k0, int hi) {
  union { v16bf v; uint4 q[2]; } f;
  f.q[0] = *(const uint4*)(row + k0 + hi * 8);
  f.q[1] = *(const uint4*)(row + k0 + 16 + hi * 8);
  return f.v;
}
// B frag: same K order as A: VGPR0-3 = K [hi*8..hi*8+7], VGPR4-7 = +16
__device__ __forceinline__ v16bf load_frag_b(const __bf16* row, size_t ofs, int hi) {
  union { v16bf v; uint4 q[2]; } f;
  f.q[0] = *(const uint4*)(row + ofs + hi * 8);
  f.q[1] = *(const uint4*)(row + ofs + 16 + hi * 8);
  return f.v;
}
#define WMMA_BF16(a, b, c) __builtin_amdgcn_wmma_f32_16x16x32_bf16(false, a, false, b, (short)0, c, false, false)

// ---------------------------------------------------------------- zero init
__global__ void zero_kernel(float* __restrict__ out, int* __restrict__ counts, size_t n) {
  size_t i = (size_t)blockIdx.x * blockDim.x + threadIdx.x;
  size_t stride = (size_t)gridDim.x * blockDim.x;
  for (size_t j = i; j < n; j += stride) out[j] = 0.0f;
  if (i < NEXP) counts[i] = 0;
}

// ---------------------------------------------------------------- x -> bf16
__global__ void __launch_bounds__(256)
convert_x_kernel(const float* __restrict__ x, uint32_t* __restrict__ xbf2, size_t n4) {
  size_t i = (size_t)blockIdx.x * blockDim.x + threadIdx.x;
  size_t stride = (size_t)gridDim.x * blockDim.x;
  for (size_t j = i; j < n4; j += stride) {
    float4 v = ((const float4*)x)[j];
    xbf2[2 * j]     = pack2bf(v.x, v.y);
    xbf2[2 * j + 1] = pack2bf(v.z, v.w);
  }
}

// ------------------------------------------------- We[e][d][h] -> bf16 Webf_t[e][h][d]
__global__ void __launch_bounds__(256)
convert_we_kernel(const float* __restrict__ We, uint32_t* __restrict__ webf2) {
  const int e  = blockIdx.z;
  const int h0 = blockIdx.x * 32;
  const int d0 = blockIdx.y * 32;
  __shared__ float tile[32][33];

  const int t = threadIdx.x;
  const float* src = We + (size_t)e * DDIM * HDIM;
#pragma unroll
  for (int i = 0; i < 4; ++i) {
    int row = (t >> 5) + i * 8;
    int ch  = t & 31;
    tile[row][ch] = src[(size_t)(d0 + row) * HDIM + h0 + ch];
  }
  __syncthreads();

  int hr  = t >> 3;
  int seg = t & 7;
  float a = tile[seg * 4 + 0][hr];
  float b = tile[seg * 4 + 1][hr];
  float c = tile[seg * 4 + 2][hr];
  float d = tile[seg * 4 + 3][hr];
  size_t base = (((size_t)e * HDIM + h0 + hr) * DDIM + d0 + seg * 4) >> 1;
  webf2[base]     = pack2bf(a, b);
  webf2[base + 1] = pack2bf(c, d);
}

// ---------------------------------------------------------------- gating
__global__ void __launch_bounds__(256)
gate_kernel(const float* __restrict__ x, const float* __restrict__ Wg,
            const float* __restrict__ bg, int* __restrict__ counts,
            int* __restrict__ idx, float* __restrict__ wgt) {
  int tok  = (blockIdx.x * blockDim.x + threadIdx.x) >> 5;
  int lane = threadIdx.x & 31;
  if (tok >= N_TOK) return;

  float acc[NEXP];
#pragma unroll
  for (int e = 0; e < NEXP; ++e) acc[e] = 0.0f;

  const float* xr = x + (size_t)tok * DDIM;
  for (int d = lane; d < DDIM; d += 32) {
    float xv = xr[d];
    const float* wr = Wg + (size_t)d * NEXP;
#pragma unroll
    for (int e = 0; e < NEXP; ++e) acc[e] += xv * wr[e];
  }
#pragma unroll
  for (int off = 16; off > 0; off >>= 1) {
#pragma unroll
    for (int e = 0; e < NEXP; ++e) acc[e] += __shfl_xor(acc[e], off, 32);
  }

  if (lane == 0) {
    float g[NEXP];
    float mx = -1e30f;
#pragma unroll
    for (int e = 0; e < NEXP; ++e) { g[e] = acc[e] + bg[e]; mx = fmaxf(mx, g[e]); }
    float s = 0.0f;
#pragma unroll
    for (int e = 0; e < NEXP; ++e) { g[e] = __expf(g[e] - mx); s += g[e]; }
    float inv = 1.0f / s;
    int e0 = 0; float v0 = -1.0f;
#pragma unroll
    for (int e = 0; e < NEXP; ++e) { g[e] *= inv; if (g[e] > v0) { v0 = g[e]; e0 = e; } }
    int e1 = -1; float v1 = -1.0f;
#pragma unroll
    for (int e = 0; e < NEXP; ++e) { if (e != e0 && g[e] > v1) { v1 = g[e]; e1 = e; } }

    int s0 = atomicAdd(&counts[e0], 1);
    idx[e0 * N_TOK + s0] = tok;  wgt[e0 * N_TOK + s0] = v0;
    int s1 = atomicAdd(&counts[e1], 1);
    idx[e1 * N_TOK + s1] = tok;  wgt[e1 * N_TOK + s1] = v1;
  }
}

// ---------------------------------------------------------------- expert GEMM (fast path)
// Wave tile: M=32 x N=64 (2x4 WMMA tiles, 8 accumulators). 12 b128 loads feed
// 8 WMMAs per K-step (0.75 KB / WMMA of L2 traffic). Block: 8 waves ->
// M=32 x N=512. Grid: (256, E, H/512).
__global__ void __launch_bounds__(256)
expert_kernel(const __bf16* __restrict__ xbf, const __bf16* __restrict__ webt,
              const float* __restrict__ be, const int* __restrict__ counts,
              const int* __restrict__ idx, const float* __restrict__ wgt,
              float* __restrict__ out) {
  const int e  = blockIdx.y;
  const int m0 = blockIdx.x * 32;
  const int hb = blockIdx.z * 512;
  const int cnt = counts[e];
  if (m0 >= cnt) return;                       // block-uniform exit

  __shared__ int   toks[32];
  __shared__ float wrow[32];
  const int t = threadIdx.x;
  if (t < 32) {
    int m = m0 + t;
    if (m < cnt) { toks[t] = idx[e * N_TOK + m]; wrow[t] = wgt[e * N_TOK + m]; }
    else         { toks[t] = -1;                 wrow[t] = 0.0f; }
  }
  __syncthreads();

  const int lane = t & 31;
  const int wv   = t >> 5;
  const int col  = lane & 15;
  const int hi   = lane >> 4;
  const int wcol = wv * 64;

  const int tkA = toks[col];
  const int tkB = toks[col + 16];
  const __bf16* arow0 = xbf + (size_t)(tkA < 0 ? 0 : tkA) * DDIM;
  const __bf16* arow1 = xbf + (size_t)(tkB < 0 ? 0 : tkB) * DDIM;
  const __bf16* brow  = webt + ((size_t)e * HDIM + hb + wcol + col) * DDIM;

  v8f acc[2][4];
#pragma unroll
  for (int i = 0; i < 2; ++i)
#pragma unroll
    for (int nt = 0; nt < 4; ++nt) acc[i][nt] = (v8f){};

  for (int k0 = 0; k0 < DDIM; k0 += BK) {
    v16bf a0 = load_frag_a(arow0, k0, hi);
    v16bf a1 = load_frag_a(arow1, k0, hi);
#pragma unroll
    for (int nt = 0; nt < 4; ++nt) {
      v16bf b = load_frag_b(brow, (size_t)nt * 16 * DDIM + k0, hi);
      acc[0][nt] = WMMA_BF16(a0, b, acc[0][nt]);
      acc[1][nt] = WMMA_BF16(a1, b, acc[1][nt]);
    }
  }

  const int hcol0 = hb + wcol + col;
  float bev[4];
#pragma unroll
  for (int nt = 0; nt < 4; ++nt) bev[nt] = be[e * HDIM + hcol0 + nt * 16];

#pragma unroll
  for (int j = 0; j < 8; ++j) {
    int mA = j + 8 * hi;                       // C VGPR j -> M = j+8*hi
    int mB = mA + 16;
    int tA = toks[mA], tB = toks[mB];
    if (tA >= 0) {
#pragma unroll
      for (int nt = 0; nt < 4; ++nt)
        atomicAdd(&out[(size_t)tA * HDIM + hcol0 + nt * 16],
                  wrow[mA] * (acc[0][nt][j] + bev[nt]));
    }
    if (tB >= 0) {
#pragma unroll
      for (int nt = 0; nt < 4; ++nt)
        atomicAdd(&out[(size_t)tB * HDIM + hcol0 + nt * 16],
                  wrow[mB] * (acc[1][nt][j] + bev[nt]));
    }
  }
}

// ------------------------------------------- fallback (small ws): on-the-fly staging
__global__ void __launch_bounds__(256)
expert_kernel_fallback(const float* __restrict__ x, const float* __restrict__ We,
                       const float* __restrict__ be, const int* __restrict__ counts,
                       const int* __restrict__ idx, const float* __restrict__ wgt,
                       float* __restrict__ out) {
  const int e  = blockIdx.y;
  const int m0 = blockIdx.x * 16;
  const int hb = blockIdx.z * BN;
  const int cnt = counts[e];
  if (m0 >= cnt) return;

  __shared__ __align__(16) __bf16 Als[16 * BK];
  __shared__ __align__(16) __bf16 Bls[BN * BK];
  __shared__ int   toks[16];
  __shared__ float wrow[16];

  const int t = threadIdx.x;
  if (t < 16) {
    int m = m0 + t;
    if (m < cnt) { toks[t] = idx[e * N_TOK + m]; wrow[t] = wgt[e * N_TOK + m]; }
    else         { toks[t] = -1;                 wrow[t] = 0.0f; }
  }
  __syncthreads();

  const int lane = t & 31;
  const int wv   = t >> 5;
  const int col  = lane & 15;
  const int hi   = lane >> 4;
  const int wcol = wv * 16;

  v8f c = {};
  const float* Wexp = We + (size_t)e * DDIM * HDIM;

  for (int k0 = 0; k0 < DDIM; k0 += BK) {
#pragma unroll
    for (int r = 0; r < 2; ++r) {
      int li  = t + r * 256;
      int row = li >> 5;
      int kk  = li & 31;
      int tk  = toks[row];
      float v = (tk >= 0) ? x[(size_t)tk * DDIM + k0 + kk] : 0.0f;
      Als[row * BK + kk] = f2bf(v);
    }
#pragma unroll
    for (int j = 0; j < 16; ++j) {
      int li = j * 256 + t;
      int hh = li & (BN - 1);
      int kk = li >> 7;
      Bls[hh * BK + kk] = f2bf(Wexp[(size_t)(k0 + kk) * HDIM + hb + hh]);
    }
    __syncthreads();

    union { v16bf v; uint4 q[2]; } af, bfr;
    af.q[0]  = *(const uint4*)&Als[col * BK + hi * 8];
    af.q[1]  = *(const uint4*)&Als[col * BK + 16 + hi * 8];
    bfr.q[0] = *(const uint4*)&Bls[(wcol + col) * BK + hi * 8];
    bfr.q[1] = *(const uint4*)&Bls[(wcol + col) * BK + 16 + hi * 8];
    c = WMMA_BF16(af.v, bfr.v, c);
    __syncthreads();
  }

  const int hcol = hb + wcol + col;
  const float bev = be[e * HDIM + hcol];
#pragma unroll
  for (int j = 0; j < 8; ++j) {
    int m  = j + 8 * hi;
    int tm = toks[m];
    if (tm >= 0) atomicAdd(&out[(size_t)tm * HDIM + hcol], wrow[m] * (c[j] + bev));
  }
}

// ---------------------------------------------------------------- launch
extern "C" void kernel_launch(void* const* d_in, const int* in_sizes, int n_in,
                              void* d_out, int out_size, void* d_ws, size_t ws_size,
                              hipStream_t stream) {
  const float* x  = (const float*)d_in[0];
  const float* Wg = (const float*)d_in[1];
  const float* bg = (const float*)d_in[2];
  const float* We = (const float*)d_in[3];
  const float* be = (const float*)d_in[4];
  float* out = (float*)d_out;

  const size_t off_idx    = 256;
  const size_t off_wgt    = off_idx + (size_t)NEXP * N_TOK * sizeof(int);
  const size_t off_xbf    = off_wgt + (size_t)NEXP * N_TOK * sizeof(float);
  const size_t xbf_bytes  = (size_t)N_TOK * DDIM * 2;
  const size_t off_webf   = off_xbf + xbf_bytes;
  const size_t webf_bytes = (size_t)NEXP * DDIM * HDIM * 2;
  const size_t need_fast  = off_webf + webf_bytes;

  int*   counts = (int*)d_ws;
  int*   idx    = (int*)((char*)d_ws + off_idx);
  float* wgt    = (float*)((char*)d_ws + off_wgt);

  zero_kernel<<<2048, 256, 0, stream>>>(out, counts, (size_t)N_TOK * HDIM);
  gate_kernel<<<N_TOK / 8, 256, 0, stream>>>(x, Wg, bg, counts, idx, wgt);

  if (ws_size >= need_fast) {
    uint32_t* xbf2  = (uint32_t*)((char*)d_ws + off_xbf);
    uint32_t* webf2 = (uint32_t*)((char*)d_ws + off_webf);

    convert_x_kernel<<<8192, 256, 0, stream>>>(x, xbf2, (size_t)N_TOK * DDIM / 4);
    dim3 tgrid(HDIM / 32, DDIM / 32, NEXP);
    convert_we_kernel<<<tgrid, 256, 0, stream>>>(We, webf2);

    dim3 grid(N_TOK / 32, NEXP, HDIM / 512);   // (256, 8, 4)
    expert_kernel<<<grid, 256, 0, stream>>>((const __bf16*)xbf2, (const __bf16*)webf2,
                                            be, counts, idx, wgt, out);
  } else {
    dim3 grid(N_TOK / 16, NEXP, HDIM / BN);    // (512, 8, 16)
    expert_kernel_fallback<<<grid, 256, 0, stream>>>(x, We, be, counts, idx, wgt, out);
  }
}
